// RNNModel_27994596836199
// MI455X (gfx1250) — matrix operand order repaired and checked by reading, not compiled, not run on hardware
//
#include <hip/hip_runtime.h>
#include <hip/hip_bf16.h>

#define T_STEPS 1024
#define B_SZ    64
#define IN_DIM  128
#define HID     512
#define OUT_DIM 128
#define NWG     8
#define COLS    (HID / NWG)      // 64 hidden columns per workgroup
#define OUTC    (OUT_DIM / NWG)  // 16 output columns per workgroup
#define THREADS 256              // 8 waves (wave32)

typedef __attribute__((ext_vector_type(16))) __bf16 v16bf;
typedef __attribute__((ext_vector_type(8)))  float  v8f;

union FragBF { v16bf v; uint4 q[2]; };
union FragF  { v8f   v; float f[8]; };

__device__ __forceinline__ unsigned short f2bf(float f) {
    unsigned int u = __float_as_uint(f);
    unsigned int r = u + 0x7FFFu + ((u >> 16) & 1u);   // round-to-nearest-even
    return (unsigned short)(r >> 16);
}
__device__ __forceinline__ float bf2f(unsigned short s) {
    return __uint_as_float(((unsigned int)s) << 16);
}

// ---------------------------------------------------------------------------
// Async-stage 64KB of bf16 hidden state: global -> LDS, bypassing VGPRs.
// GLOBAL_LOAD_ASYNC_TO_LDS_B128 (GVS mode), tracked by ASYNCcnt.
// scope:SCOPE_DEV: h slices were produced by other WGPs last step and this
// LDS buffer alternates between two global buffers, so stale WGP$ lines must
// be bypassed.
// ---------------------------------------------------------------------------
__device__ __forceinline__ void stage_h_async(unsigned short* sH,
                                              const unsigned short* src_glb,
                                              int tix) {
    const unsigned lds_base = (unsigned)(unsigned long long)(const void*)sH;
#pragma unroll
    for (int i = 0; i < (B_SZ * HID * 2) / 16 / THREADS; ++i) {   // 16 x b128/thread
        const unsigned off = (unsigned)((tix + i * THREADS) * 16);
        asm volatile("global_load_async_to_lds_b128 %0, %1, %2 scope:SCOPE_DEV"
                     :
                     : "v"(lds_base + off), "v"(off), "s"(src_glb)
                     : "memory");
    }
    asm volatile("s_wait_asynccnt 0x0" ::: "memory");
}

// ---------------------------------------------------------------------------
// Init kernel: zero the global barrier, convert h0 (f32) -> hbuf[0] (bf16).
// Re-run every launch so persistent-kernel state is deterministic per call.
// ---------------------------------------------------------------------------
__global__ void rnn_init(const float* __restrict__ h0,
                         unsigned short* __restrict__ hbuf,
                         unsigned int* __restrict__ bar) {
    if (blockIdx.x == 0 && threadIdx.x < 8) bar[threadIdx.x] = 0u;
    int stride = gridDim.x * blockDim.x;
    for (int i = blockIdx.x * blockDim.x + threadIdx.x; i < B_SZ * HID; i += stride)
        hbuf[i] = f2bf(h0[i]);
}

// ---------------------------------------------------------------------------
// y[t] = r_t @ W_fc.T + b_fc  (this WG's 16 output columns), r_t staged in sH.
// Waves 0..3 each own one 16-row M tile of the [64 x 16] result.
// ---------------------------------------------------------------------------
__device__ __forceinline__ void y_tile_compute(
    const unsigned short* __restrict__ sH,    // [64][512] bf16
    const unsigned short* __restrict__ sWfc,  // [16][512] bf16 (row = out col)
    const float* __restrict__ bfc,            // [16]
    float* __restrict__ yp,                   // -> y[t][0][0]
    int g, int wave, int half, int ln16)
{
    if (wave >= 4) return;
    const int tm  = wave;
    const int akh = half ? 16 : 0;     // A-matrix 16x32 lane K-offset (contiguous, matches B)
    const int bkh = half ? 16 : 0;     // B-matrix 32x16 lane K-offset
    FragF acc;
    float bb = bfc[ln16];
#pragma unroll
    for (int i = 0; i < 8; ++i) acc.f[i] = bb;
#pragma unroll
    for (int k0 = 0; k0 < HID; k0 += 32) {
        FragBF a, b;
        const int am = tm * 16 + ln16;
        a.q[0] = *(const uint4*)&sH[am * HID + k0 + akh];
        a.q[1] = *(const uint4*)&sH[am * HID + k0 + akh + 8];
        b.q[0] = *(const uint4*)&sWfc[ln16 * HID + k0 + bkh];
        b.q[1] = *(const uint4*)&sWfc[ln16 * HID + k0 + bkh + 8];
        acc.v = __builtin_amdgcn_wmma_f32_16x16x32_bf16(
            false, a.v, false, b.v, (short)0, acc.v, false, false);
    }
#pragma unroll
    for (int i = 0; i < 8; ++i) {
        int m = tm * 16 + i + (half ? 8 : 0);          // C/D layout: VGPR i -> M=i / M=8+i
        yp[(size_t)m * OUT_DIM + g * OUTC + ln16] = acc.f[i];
    }
}

// ---------------------------------------------------------------------------
// Persistent fused RNN kernel. Grid = NWG blocks, each pinned to one WGP.
// ---------------------------------------------------------------------------
__global__ void __launch_bounds__(THREADS, 1)
rnn_persistent(const float* __restrict__ x,
               const float* __restrict__ W_ih, const float* __restrict__ W_hh,
               const float* __restrict__ b_ih, const float* __restrict__ b_hh,
               const float* __restrict__ W_fc, const float* __restrict__ b_fc,
               float* __restrict__ y_out, float* __restrict__ hlast_out,
               unsigned short* __restrict__ hbuf,   // [2][B][HID] bf16 double buffer
               unsigned int* __restrict__ bar)
{
    extern __shared__ unsigned char smem_raw[];
    unsigned short* sH   = (unsigned short*)smem_raw;   // [64][512]  h_{t} staging
    unsigned short* sX   = sH   + B_SZ * HID;           // [64][128]  x_t  (bf16)
    unsigned short* sWhh = sX   + B_SZ * IN_DIM;        // [64][512]  W_hh slice
    unsigned short* sWih = sWhh + COLS * HID;           // [64][128]  W_ih slice
    unsigned short* sWfc = sWih + COLS * IN_DIM;        // [16][512]  W_fc slice
    float* bcomb = (float*)(sWfc + OUTC * HID);         // [64] b_ih+b_hh slice
    float* bfc   = bcomb + COLS;                        // [16]

    const int g    = blockIdx.x;
    const int tix  = threadIdx.x;
    const int wave = tix >> 5;
    const int lane = tix & 31;
    const int half = lane >> 4;     // 0: lanes 0-15, 1: lanes 16-31
    const int ln16 = lane & 15;
    const int akh  = half ? 16 : 0;
    const int bkh  = half ? 16 : 0;

    // Tile pairing: wave w owns C-tiles 2w and 2w+1 of the [64x64] h_new slice.
    // Both share the same M-tile -> A fragment loaded ONCE for two WMMAs.
    const int tm  = (wave * 2) >> 2;   // 0..3
    const int tn0 = (wave * 2) & 3;    // 0 or 2
    const int tn1 = tn0 + 1;

    // ---- one-time weight preload into LDS (resident for all 1024 steps) ----
    for (int i = tix; i < COLS * HID; i += THREADS) {
        int r = i / HID, c = i % HID;
        sWhh[i] = f2bf(W_hh[(size_t)(g * COLS + r) * HID + c]);
    }
    for (int i = tix; i < COLS * IN_DIM; i += THREADS) {
        int r = i / IN_DIM, c = i % IN_DIM;
        sWih[i] = f2bf(W_ih[(size_t)(g * COLS + r) * IN_DIM + c]);
    }
    for (int i = tix; i < OUTC * HID; i += THREADS) {
        int r = i / HID, c = i % HID;
        sWfc[i] = f2bf(W_fc[(size_t)(g * OUTC + r) * HID + c]);
    }
    for (int i = tix; i < COLS; i += THREADS) bcomb[i] = b_ih[g * COLS + i] + b_hh[g * COLS + i];
    for (int i = tix; i < OUTC; i += THREADS) bfc[i] = b_fc[g * OUTC + i];
    __syncthreads();

#pragma unroll 1
    for (int t = 0; t < T_STEPS; ++t) {
        // ---- async-stage h_t (= r[t-1], bf16) straight into LDS ----
        stage_h_async(sH, hbuf + (size_t)(t & 1) * B_SZ * HID, tix);
        // ---- stage x_t with f32->bf16 conversion ----
        {
            const float* xt = x + (size_t)t * B_SZ * IN_DIM;
            for (int i = tix; i < B_SZ * IN_DIM; i += THREADS) sX[i] = f2bf(xt[i]);
        }
        // prefetch next step's input while this step computes
        if (t + 1 < T_STEPS)
            __builtin_prefetch(x + (size_t)(t + 1) * B_SZ * IN_DIM + tix * 32, 0, 3);
        __syncthreads();

        // ---- pipelined output projection: y[t-1] = r[t-1] @ W_fc.T + b_fc ----
        if (t > 0)
            y_tile_compute(sH, sWfc, bfc, y_out + (size_t)(t - 1) * B_SZ * OUT_DIM,
                           g, wave, half, ln16);

        // ---- h_new[:, g-slice] = tanh(x_t@W_ih.T + h_t@W_hh.T + b) ----
        unsigned short* hdst = hbuf + (size_t)((t + 1) & 1) * B_SZ * HID;
        {
            FragF acc0, acc1;
            const float bb0 = bcomb[tn0 * 16 + ln16];
            const float bb1 = bcomb[tn1 * 16 + ln16];
#pragma unroll
            for (int i = 0; i < 8; ++i) { acc0.f[i] = bb0; acc1.f[i] = bb1; }
            const int am = tm * 16 + ln16;
            // input projection, K over IN_DIM (shared A, two B streams)
#pragma unroll
            for (int k0 = 0; k0 < IN_DIM; k0 += 32) {
                FragBF a, b0, b1;
                a.q[0]  = *(const uint4*)&sX[am * IN_DIM + k0 + akh];
                a.q[1]  = *(const uint4*)&sX[am * IN_DIM + k0 + akh + 8];
                b0.q[0] = *(const uint4*)&sWih[(tn0 * 16 + ln16) * IN_DIM + k0 + bkh];
                b0.q[1] = *(const uint4*)&sWih[(tn0 * 16 + ln16) * IN_DIM + k0 + bkh + 8];
                b1.q[0] = *(const uint4*)&sWih[(tn1 * 16 + ln16) * IN_DIM + k0 + bkh];
                b1.q[1] = *(const uint4*)&sWih[(tn1 * 16 + ln16) * IN_DIM + k0 + bkh + 8];
                acc0.v = __builtin_amdgcn_wmma_f32_16x16x32_bf16(
                    false, a.v, false, b0.v, (short)0, acc0.v, false, false);
                acc1.v = __builtin_amdgcn_wmma_f32_16x16x32_bf16(
                    false, a.v, false, b1.v, (short)0, acc1.v, false, false);
            }
            // recurrent projection, K over HID (shared A, two B streams)
#pragma unroll
            for (int k0 = 0; k0 < HID; k0 += 32) {
                FragBF a, b0, b1;
                a.q[0]  = *(const uint4*)&sH[am * HID + k0 + akh];
                a.q[1]  = *(const uint4*)&sH[am * HID + k0 + akh + 8];
                b0.q[0] = *(const uint4*)&sWhh[(tn0 * 16 + ln16) * HID + k0 + bkh];
                b0.q[1] = *(const uint4*)&sWhh[(tn0 * 16 + ln16) * HID + k0 + bkh + 8];
                b1.q[0] = *(const uint4*)&sWhh[(tn1 * 16 + ln16) * HID + k0 + bkh];
                b1.q[1] = *(const uint4*)&sWhh[(tn1 * 16 + ln16) * HID + k0 + bkh + 8];
                acc0.v = __builtin_amdgcn_wmma_f32_16x16x32_bf16(
                    false, a.v, false, b0.v, (short)0, acc0.v, false, false);
                acc1.v = __builtin_amdgcn_wmma_f32_16x16x32_bf16(
                    false, a.v, false, b1.v, (short)0, acc1.v, false, false);
            }
            // tanh + publish h_new slices (bf16) to the global double buffer
#pragma unroll
            for (int i = 0; i < 8; ++i) {
                const int m = tm * 16 + i + (half ? 8 : 0);
                hdst[(size_t)m * HID + g * COLS + tn0 * 16 + ln16] = f2bf(tanhf(acc0.f[i]));
                hdst[(size_t)m * HID + g * COLS + tn1 * 16 + ln16] = f2bf(tanhf(acc1.f[i]));
            }
        }

        // ---- device-wide step barrier (monotonic counter, init'd per launch) ----
        __syncthreads();
        if (tix == 0) {
            __threadfence();
            __hip_atomic_fetch_add(bar, 1u, __ATOMIC_ACQ_REL, __HIP_MEMORY_SCOPE_AGENT);
            const unsigned int target = (unsigned int)(t + 1) * NWG;
            while (__hip_atomic_load(bar, __ATOMIC_ACQUIRE, __HIP_MEMORY_SCOPE_AGENT) < target)
                __builtin_amdgcn_s_sleep(2);
        }
        __syncthreads();
    }

    // ---- epilogue: y[T-1] and h_last from r[T-1] ----
    stage_h_async(sH, hbuf + (size_t)(T_STEPS & 1) * B_SZ * HID, tix);
    __syncthreads();
    y_tile_compute(sH, sWfc, bfc, y_out + (size_t)(T_STEPS - 1) * B_SZ * OUT_DIM,
                   g, wave, half, ln16);
    for (int i = tix; i < B_SZ * COLS; i += THREADS) {
        int m = i / COLS, c = i % COLS;
        hlast_out[(size_t)m * HID + g * COLS + c] = bf2f(sH[m * HID + g * COLS + c]);
    }
}

// ---------------------------------------------------------------------------
extern "C" void kernel_launch(void* const* d_in, const int* in_sizes, int n_in,
                              void* d_out, int out_size, void* d_ws, size_t ws_size,
                              hipStream_t stream) {
    const float* x    = (const float*)d_in[0];
    const float* h0   = (const float*)d_in[1];
    const float* W_ih = (const float*)d_in[2];
    const float* W_hh = (const float*)d_in[3];
    const float* b_ih = (const float*)d_in[4];
    const float* b_hh = (const float*)d_in[5];
    const float* W_fc = (const float*)d_in[6];
    const float* b_fc = (const float*)d_in[7];

    float* y_out = (float*)d_out;
    float* hlast = y_out + (size_t)T_STEPS * B_SZ * OUT_DIM;

    unsigned int*   bar  = (unsigned int*)d_ws;
    unsigned short* hbuf = (unsigned short*)((char*)d_ws + 256);

    const size_t smem =
        (size_t)(B_SZ * HID + B_SZ * IN_DIM + COLS * HID + COLS * IN_DIM + OUTC * HID) *
            sizeof(unsigned short) +
        (size_t)(COLS + OUTC) * sizeof(float);   // ~176.3 KB, under 320 KB/WGP

    (void)hipFuncSetAttribute((const void*)rnn_persistent,
                              hipFuncAttributeMaxDynamicSharedMemorySize, (int)smem);

    rnn_init<<<64, 256, 0, stream>>>(h0, hbuf, bar);
    rnn_persistent<<<NWG, THREADS, smem, stream>>>(x, W_ih, W_hh, b_ih, b_hh,
                                                   W_fc, b_fc, y_out, hlast, hbuf, bar);
}
